// ContextInteractionModel_26096221290655
// MI455X (gfx1250) — compile-verified, matrix operand order checked by reading, not run on hardware
//
#include <hip/hip_runtime.h>

// ---- CDNA5 WMMA types -------------------------------------------------------
typedef __attribute__((ext_vector_type(16))) __bf16       v16bf;
typedef __attribute__((ext_vector_type(8)))  float        v8f;
typedef __attribute__((ext_vector_type(4)))  unsigned int v4u;

#define BB 512   // batch
#define LL 128   // context length
#define DD 512   // embed dim

// Load a 16x32 bf16 WMMA A/B fragment from a row-major (stride DD) matrix.
// ISA 16-bit A layout: lanes 0-15 -> row M=lane, K = {0..7, 16..23};
// lanes 16-31 -> row M=lane-16, K = {8..15, 24..31}. Two contiguous 16B chunks.
union FragU { v16bf v; v4u q[2]; };

__device__ __forceinline__ v16bf load_frag(const __bf16* base, int rowBase,
                                           int kBase, int lane) {
  const int r  = rowBase + (lane & 15);
  const int k0 = kBase + ((lane & 16) ? 8 : 0);
  const __bf16* p = base + r * DD + k0;
  FragU u;
  u.q[0] = *(const v4u*)(p);
  u.q[1] = *(const v4u*)(p + 16);
  return u.v;
}

// ---- kernel 1: att_mat f32 (K-major) -> bf16 N-major (Wt[n][k]) -------------
__global__ void __launch_bounds__(256)
cim_wconv(const float* __restrict__ W, __bf16* __restrict__ Wt) {
  const int idx = blockIdx.x * 256 + threadIdx.x;   // 0 .. 512*512-1
  const int n = idx >> 9, k = idx & 511;
  Wt[idx] = (__bf16)W[k * DD + n];
}

// ---- kernel 2: gather embeddings -> bf16 staging + sum-of-squares ----------
__global__ void __launch_bounds__(256)
cim_gather(const int* __restrict__ t1c, const int* __restrict__ t2c,
           const float* __restrict__ emb,
           __bf16* __restrict__ Abf, __bf16* __restrict__ Bbf,
           float* __restrict__ ssA, float* __restrict__ ssB) {
  const int b = blockIdx.x, tid = threadIdx.x;
  float sA = 0.f, sB = 0.f;
  for (int l = 0; l < LL; ++l) {
    const int r1 = t1c[b * LL + l];
    const int r2 = t2c[b * LL + l];
    const size_t o = (size_t)(b * LL + l) * DD;
    for (int d = tid; d < DD; d += 256) {
      float v = emb[(size_t)r1 * DD + d];
      float w = emb[(size_t)r2 * DD + d];
      Abf[o + d] = (__bf16)v;  sA += v * v;
      Bbf[o + d] = (__bf16)w;  sB += w * w;
    }
  }
  __shared__ float r0[256], r1s[256];
  r0[tid] = sA; r1s[tid] = sB;
  __syncthreads();
  for (int s = 128; s > 0; s >>= 1) {
    if (tid < s) { r0[tid] += r0[tid + s]; r1s[tid] += r1s[tid + s]; }
    __syncthreads();
  }
  if (tid == 0) { ssA[b] = r0[0]; ssB[b] = r1s[0]; }
}

// ---- kernel 3: per-batch WMMA pipeline -------------------------------------
__global__ void __launch_bounds__(256)
cim_main(const __bf16* __restrict__ Wt, const __bf16* __restrict__ Abf,
         const __bf16* __restrict__ Bbf, const float* __restrict__ ssA,
         const float* __restrict__ ssB, const float* __restrict__ w_pred,
         const float* __restrict__ b_pred, float* __restrict__ out) {
  extern __shared__ __align__(16) char smem[];
  __bf16* A_s = (__bf16*)smem;              // 128x512 bf16 (128KB); later reused as S f32 (64KB)
  __bf16* T_s = (__bf16*)(smem + 131072);   // 128x512 bf16 (128KB)
  float* rowsum = (float*)(smem + 262144);  // 128
  float* colsum = rowsum + 128;             // 128
  float* rowsP  = colsum + 128;             // 128
  float* colsP  = rowsP + 128;              // 128
  float* nA_s   = colsP + 128;              // 512
  float* nB_s   = nA_s + 512;               // 512
  float* red    = nB_s + 512;               // 256

  const int b = blockIdx.x;
  const int tid = threadIdx.x;
  const int wave = tid >> 5, lane = tid & 31;

  const __bf16* Ag = Abf + (size_t)b * LL * DD;
  const __bf16* Bg = Bbf + (size_t)b * LL * DD;
  const float sa = ssA[b], sb = ssB[b];
  const float scale = rsqrtf(sa * sb);      // 1/(||A||_F * ||B||_F)
  const float invnA = rsqrtf(sa), invnB = rsqrtf(sb);

  __builtin_prefetch(Wt + tid * 128, 0, 1); // warm W path (global_prefetch_b8)

  // stage A tile into LDS (128KB) via the CDNA5 async memory->LDS path.
  // Dynamic LDS is the only LDS in this kernel, so A_s lives at LDS offset 0
  // and the per-lane VDST VGPR can carry the raw byte offset directly.
  {
    const unsigned char* gp = (const unsigned char*)Ag;
    for (int i = tid; i < (LL * DD * 2) / 16; i += 256) {
      const unsigned lds_off = (unsigned)i * 16u;
      const unsigned char* ga = gp + (size_t)i * 16u;
      asm volatile("global_load_async_to_lds_b128 %0, %1, off"
                   :: "v"(lds_off), "v"(ga) : "memory");
    }
    asm volatile("s_wait_asynccnt 0" ::: "memory");
  }
  __syncthreads();

  // GEMM1: T(128x512) = A(128x512, LDS) x W(512x512, bf16 N-major in L2)
  // 8x32 = 256 output tiles of 16x16, 8 waves, K-loop = 512/32 = 16
  for (int t = wave; t < 256; t += 8) {
    const int mT = t >> 5, nT = t & 31;
    v8f acc = {0.f, 0.f, 0.f, 0.f, 0.f, 0.f, 0.f, 0.f};
    for (int kk = 0; kk < 16; ++kk) {
      v16bf af = load_frag(A_s, mT * 16, kk * 32, lane);
      v16bf bf = load_frag(Wt,  nT * 16, kk * 32, lane);
      acc = __builtin_amdgcn_wmma_f32_16x16x32_bf16(false, af, false, bf,
                                                    (short)0, acc, false, false);
    }
    const int nn = nT * 16 + (lane & 15);
    const int mB = mT * 16 + ((lane & 16) ? 8 : 0);
    for (int r = 0; r < 8; ++r)
      T_s[(mB + r) * DD + nn] = (__bf16)acc[r];
  }
  __syncthreads();

  float* S_s = (float*)smem;                     // reuse A_s region (64KB)
  float* Sg  = out + BB + (size_t)b * LL * LL;   // output tuple: [logits | S]

  // GEMM2: S(128x128) = T(LDS) x Braw^T; Braw rows are B-fragment columns
  for (int t = wave; t < 64; t += 8) {
    const int mT = t >> 3, nT = t & 7;
    v8f acc = {0.f, 0.f, 0.f, 0.f, 0.f, 0.f, 0.f, 0.f};
    for (int kk = 0; kk < 16; ++kk) {
      v16bf af = load_frag(T_s, mT * 16, kk * 32, lane);
      v16bf bf = load_frag(Bg,  nT * 16, kk * 32, lane);
      acc = __builtin_amdgcn_wmma_f32_16x16x32_bf16(false, af, false, bf,
                                                    (short)0, acc, false, false);
    }
    const int nn = nT * 16 + (lane & 15);
    const int mB = mT * 16 + ((lane & 16) ? 8 : 0);
    for (int r = 0; r < 8; ++r) {
      const int mm = mB + r;
      const float sv = tanhf(acc[r] * scale);
      S_s[mm * LL + nn] = sv;
      Sg[mm * LL + nn]  = sv;
    }
  }
  __syncthreads();

  // deterministic row/col sums from LDS S (no float atomics)
  if (tid < 128) {
    float rs = 0.f, cs = 0.f;
    for (int j = 0; j < 128; ++j) {
      rs += S_s[tid * LL + j];
      cs += S_s[j * LL + tid];
    }
    rowsum[tid] = rs; colsum[tid] = cs;
  }
  __syncthreads();

  // softmax over means (mean = sum/128)
  if (tid < 128) {
    float mr = -1e30f, mc = -1e30f;
    for (int j = 0; j < 128; ++j) {
      mr = fmaxf(mr, rowsum[j]);
      mc = fmaxf(mc, colsum[j]);
    }
    float er = 0.f, ec = 0.f;
    for (int j = 0; j < 128; ++j) {
      er += __expf((rowsum[j] - mr) * (1.f / 128.f));
      ec += __expf((colsum[j] - mc) * (1.f / 128.f));
    }
    rowsP[tid] = __expf((rowsum[tid] - mr) * (1.f / 128.f)) / er;
    colsP[tid] = __expf((colsum[tid] - mc) * (1.f / 128.f)) / ec;
  }
  __syncthreads();

  // newA[d] = sum_l rows[l]*A[l,d]/||A||, newB likewise (A/B re-read from L2)
  for (int d = tid; d < DD; d += 256) {
    float a = 0.f, c = 0.f;
    for (int l = 0; l < LL; ++l) {
      a += rowsP[l] * (float)Ag[l * DD + d];
      c += colsP[l] * (float)Bg[l * DD + d];
    }
    nA_s[d] = a * invnA;
    nB_s[d] = c * invnB;
  }
  __syncthreads();

  // logits[b] = dot(newA*newB, w_pred) + b_pred
  float part = 0.f;
  for (int d = tid; d < DD; d += 256)
    part += nA_s[d] * nB_s[d] * w_pred[d];
  red[tid] = part;
  __syncthreads();
  for (int s = 128; s > 0; s >>= 1) {
    if (tid < s) red[tid] += red[tid + s];
    __syncthreads();
  }
  if (tid == 0) out[b] = red[0] + b_pred[0];
}

// ---- launcher ---------------------------------------------------------------
extern "C" void kernel_launch(void* const* d_in, const int* in_sizes, int n_in,
                              void* d_out, int out_size, void* d_ws, size_t ws_size,
                              hipStream_t stream) {
  (void)in_sizes; (void)n_in; (void)out_size; (void)ws_size;
  // inputs: 0:t1s 1:t2s (dead code in ref) 2:t1_contexts 3:t2_contexts
  //         4:embed_table 5:att_mat 6:w_pred 7:b_pred
  const int*   t1c = (const int*)d_in[2];
  const int*   t2c = (const int*)d_in[3];
  const float* emb = (const float*)d_in[4];
  const float* att = (const float*)d_in[5];
  const float* wp  = (const float*)d_in[6];
  const float* bp  = (const float*)d_in[7];
  float* out = (float*)d_out;

  // workspace layout
  char* ws = (char*)d_ws;
  __bf16* Wt  = (__bf16*)ws;                               // 512KB  (bf16 W, N-major)
  __bf16* Abf = (__bf16*)(ws + 524288);                    // 64MB
  __bf16* Bbf = (__bf16*)(ws + 524288 + 67108864);         // 64MB
  float*  ssA = (float*)(ws + 524288 + 2ll * 67108864);    // 2KB
  float*  ssB = ssA + BB;                                  // 2KB

  cim_wconv <<<(DD * DD) / 256, 256, 0, stream>>>(att, Wt);
  cim_gather<<<BB, 256, 0, stream>>>(t1c, t2c, emb, Abf, Bbf, ssA, ssB);
  // dynamic LDS: 2*128KB bf16 tiles + 7168B f32 scratch = 269312B (<320KB/WGP)
  cim_main  <<<BB, 256, 269312, stream>>>(Wt, Abf, Bbf, ssA, ssB, wp, bp, out);
}
